// TN_60997125538288
// MI455X (gfx1250) — matrix operand order repaired and checked by reading, not compiled, not run on hardware
//
#include <hip/hip_runtime.h>

typedef __attribute__((ext_vector_type(16))) _Float16 v16h;
typedef __attribute__((ext_vector_type(8)))  _Float16 v8h;
typedef __attribute__((ext_vector_type(2)))  __fp16   p2h;   // cvt_pkrtz result type
typedef __attribute__((ext_vector_type(8)))  float    v8f;

#define S_LEN 1024
#define RDIM  32

// Load the 32 f32 elements this lane needs for B0/B1 of one step:
// lane half h supplies K rows 8h..8h+7 and 16+8h..16+8h+7, column n (B0) and n+16 (B1).
#define LOAD_M(dst, tt) do {                                                    \
    const float* gp_ = G + (size_t)toks[(tt)] * (RDIM * RDIM)                   \
                         + (size_t)((h << 3) * RDIM) + (size_t)n;               \
    _Pragma("unroll")                                                           \
    for (int k_ = 0; k_ < 16; ++k_) {                                           \
      dst[k_]      = gp_[(k_ + (k_ >= 8 ? 8 : 0)) * RDIM];                      \
      dst[16 + k_] = gp_[(k_ + (k_ >= 8 ? 8 : 0)) * RDIM + 16];                 \
    }                                                                           \
  } while (0)

// One chain step: B = f16(M), D = A*B (two 16x16 halves), y = row0(D),
// s = sign(y), rebuild replicated-row A from s via LDS.
// sign: for |y| >= eps the reference y/max(|y|,eps) is exactly copysign(1,y);
// for |y| < eps the value is <= 1e-12/1e-12-scale ~ 0 and flushes in f16 -> 0.
#define STEP(raw) do {                                                          \
    v16h B0, B1;                                                                \
    _Pragma("unroll")                                                           \
    for (int vv_ = 0; vv_ < 8; ++vv_) {                                         \
      p2h p0_ = __builtin_amdgcn_cvt_pkrtz(raw[2*vv_],      raw[2*vv_ + 1]);    \
      p2h p1_ = __builtin_amdgcn_cvt_pkrtz(raw[16 + 2*vv_], raw[16 + 2*vv_+1]); \
      B0[2*vv_] = (_Float16)p0_[0]; B0[2*vv_ + 1] = (_Float16)p0_[1];           \
      B1[2*vv_] = (_Float16)p1_[0]; B1[2*vv_ + 1] = (_Float16)p1_[1];           \
    }                                                                           \
    v8f z_ = {};                                                                \
    v8f d0_ = __builtin_amdgcn_wmma_f32_16x16x32_f16(                           \
        false, A, false, B0, (short)0, z_, false, false);                       \
    v8f d1_ = __builtin_amdgcn_wmma_f32_16x16x32_f16(                           \
        false, A, false, B1, (short)0, z_, false, false);                       \
    float yv_ = (lane < 16) ? d0_[0] : d1_[0];                                  \
    float sg_ = __builtin_copysignf(1.0f, yv_);                                 \
    sv = (fabsf(yv_) >= 1e-12f) ? sg_ : 0.0f;                                   \
    ysh[lane] = (_Float16)sv;                                                   \
    __builtin_amdgcn_wave_barrier();                                            \
    v8h lo_ = *(const v8h*)(ysh + 8 * h);                                       \
    v8h hi_ = *(const v8h*)(ysh + 16 + 8 * h);                                  \
    _Pragma("unroll")                                                           \
    for (int j_ = 0; j_ < 8; ++j_) { A[j_] = lo_[j_]; A[8 + j_] = hi_[j_]; }    \
    __builtin_amdgcn_wave_barrier();                                            \
  } while (0)

__launch_bounds__(32, 1)
__global__ void sign_chain_kernel(const int*   __restrict__ x,
                                  const float* __restrict__ G,
                                  const float* __restrict__ G1,
                                  const float* __restrict__ lw,
                                  const float* __restrict__ lb,
                                  float*       __restrict__ out)
{
  const int b    = blockIdx.x;
  const int lane = threadIdx.x;     // 0..31, wave32
  const int h    = lane >> 4;       // K-half this lane serves
  const int n    = lane & 15;       // column within half

  __shared__ __attribute__((aligned(16))) int      toks[S_LEN];
  __shared__ __attribute__((aligned(16))) _Float16 ysh[RDIM];

  // Stage this batch's 1024 token ids into LDS (8 x int4 per lane).
  {
    const int4* xr = (const int4*)(x + (size_t)b * S_LEN);
    int4* td = (int4*)toks;
    #pragma unroll
    for (int i = 0; i < S_LEN / 4 / 32; ++i)
      td[lane + 32 * i] = xr[lane + 32 * i];
  }
  __builtin_amdgcn_wave_barrier();   // single wave: DS is HW-ordered in program order

  // Initial vector v = G1_weight (no sign applied before first step).
  ysh[lane] = (_Float16)G1[lane];
  __builtin_amdgcn_wave_barrier();

  // Replicated-row A: all 16 rows equal v. Per-lane layout:
  // elems 0..7  = v[8h .. 8h+7], elems 8..15 = v[16+8h .. 16+8h+7].
  v16h A;
  {
    v8h lo = *(const v8h*)(ysh + 8 * h);
    v8h hi = *(const v8h*)(ysh + 16 + 8 * h);
    #pragma unroll
    for (int j = 0; j < 8; ++j) { A[j] = lo[j]; A[8 + j] = hi[j]; }
  }

  float raw0[32], raw1[32];
  float sv = 0.0f;

  LOAD_M(raw0, 0);

  for (int t = 0; t < S_LEN; t += 2) {
    // Warm L2 for step t+4's matrix: one 128B row per lane = full 4KB tile.
    {
      int tp = (t + 4 < S_LEN) ? (t + 4) : (S_LEN - 1);
      const float* gpf = G + (size_t)toks[tp] * (RDIM * RDIM);
      __builtin_prefetch(gpf + lane * RDIM, 0, 0);
    }

    LOAD_M(raw1, t + 1);        // prefetch next step's matrix into VGPRs
    STEP(raw0);                 // dependent chain step t

    int t2 = (t + 2 < S_LEN) ? (t + 2) : (S_LEN - 1);
    LOAD_M(raw0, t2);
    STEP(raw1);                 // dependent chain step t+1
  }

  // out[b] = sum_j v_j * lin_w[j] + lin_b  (wave32 butterfly reduction)
  float c = sv * lw[lane];
  #pragma unroll
  for (int off = 16; off > 0; off >>= 1)
    c += __shfl_xor(c, off, 32);
  if (lane == 0) out[b] = c + lb[0];
}

extern "C" void kernel_launch(void* const* d_in, const int* in_sizes, int n_in,
                              void* d_out, int out_size, void* d_ws, size_t ws_size,
                              hipStream_t stream)
{
  const int*   x  = (const int*)  d_in[0];
  const float* G  = (const float*)d_in[1];
  const float* G1 = (const float*)d_in[2];
  const float* lw = (const float*)d_in[3];
  const float* lb = (const float*)d_in[4];
  float* out = (float*)d_out;

  sign_chain_kernel<<<dim3(64), dim3(32), 0, stream>>>(x, G, G1, lw, lb, out);
}
